// SAGEEncoderBlock_6098853560618
// MI455X (gfx1250) — matrix operand order repaired and checked by reading, not compiled, not run on hardware
//
#include <hip/hip_runtime.h>
#include <hip/hip_bf16.h>

// ---------------------------------------------------------------------------
// GraphSAGE 2-layer encoder for MI455X (gfx1250, wave32).
//  - scatter/gather phase: f32 global atomics into accumulators that fit the
//    192MB L2 (51MB / 102MB), one wave per edge, float4 lanes.
//  - dense phase: v_wmma_f32_16x16x32_bf16, one 16x16 tile per wave, fused
//    mean-scale + bias + ReLU + residual epilogue.
// ---------------------------------------------------------------------------

typedef __attribute__((ext_vector_type(16))) __bf16 v16bf;
typedef __attribute__((ext_vector_type(8)))  float  v8f;

#define N_NODES 100000
#define N_EDGES 3200000

static __device__ inline v8f wmma_bf16(v16bf a, v16bf b, v8f c) {
    // (neg_a, A, neg_b, B, c_mod, C, reuse_a, reuse_b)
    return __builtin_amdgcn_wmma_f32_16x16x32_bf16(false, a, false, b,
                                                   (short)0, c, false, false);
}

// Build 16x32 bf16 A-fragment for one lane from an fp32 row (row-major),
// optionally scaled. Lane half=lane>>4 holds
// K = [16*half .. 16*half+15] (contiguous) of row M=lane&15.
static __device__ inline v16bf load_a_frag(const float* __restrict__ row,
                                           int half, float scale) {
    const float4* p0 = (const float4*)(row + 16 * half);
    const float4* p1 = (const float4*)(row + 16 * half + 8);
    float4 c0 = p0[0], c1 = p0[1];
    float4 c2 = p1[0], c3 = p1[1];
    float t[16] = {c0.x, c0.y, c0.z, c0.w, c1.x, c1.y, c1.z, c1.w,
                   c2.x, c2.y, c2.z, c2.w, c3.x, c3.y, c3.z, c3.w};
    v16bf a;
#pragma unroll
    for (int i = 0; i < 16; ++i) a[i] = (__bf16)(t[i] * scale);
    return a;
}

// ----------------------------- small utility kernels -----------------------

__global__ void cvt_bf16_kernel(const float* __restrict__ w,
                                __bf16* __restrict__ o, int n) {
    int i = blockIdx.x * blockDim.x + threadIdx.x;
    if (i < n) o[i] = (__bf16)w[i];
}

__global__ void count_kernel(const long long* __restrict__ dst,
                             float* __restrict__ cnt, int nE) {
    int e = blockIdx.x * blockDim.x + threadIdx.x;
    if (e < nE) atomicAdd(&cnt[(int)dst[e]], 1.0f);
}

__global__ void inv_kernel(float* __restrict__ cnt, int n) {
    int i = blockIdx.x * blockDim.x + threadIdx.x;
    if (i < n) cnt[i] = 1.0f / fmaxf(cnt[i], 1.0f);
}

// One wave per edge; lanes cover C floats with float4 loads + f32 atomics.
// Accumulator (<=102MB) lives in L2 (192MB), so atomics stay on-chip.
template <int C>
__global__ void __launch_bounds__(256)
scatter_kernel(const float* __restrict__ x, const long long* __restrict__ src,
               const long long* __restrict__ dst, float* __restrict__ agg,
               int nE) {
    int gid  = blockIdx.x * blockDim.x + threadIdx.x;
    int e    = gid >> 5;
    int lane = gid & 31;
    if (e >= nE) return;
    int s = (int)src[e];
    int d = (int)dst[e];
    const float4* xs = (const float4*)(x + (size_t)s * C) + lane;
    float* ad = agg + (size_t)d * C + lane * 4;
#pragma unroll
    for (int i = 0; i < C / 128; ++i) {
        float4 v = xs[32 * i];
        atomicAdd(ad + 128 * i + 0, v.x);
        atomicAdd(ad + 128 * i + 1, v.y);
        atomicAdd(ad + 128 * i + 2, v.z);
        atomicAdd(ad + 128 * i + 3, v.w);
    }
}

// ----------------------------- fused WMMA layers ---------------------------
// Layer 1: h1 = relu(mean @ Wl^T + x @ Wr^T + b) + x @ Ws^T   (128 -> 256)
__global__ void __launch_bounds__(256)
gemm_layer1(const float* __restrict__ agg, const float* __restrict__ inv_cnt,
            const float* __restrict__ x, const __bf16* __restrict__ Wl,
            const __bf16* __restrict__ Wr, const __bf16* __restrict__ Ws,
            const float* __restrict__ bias, float* __restrict__ h1) {
    const int Cin = 128, Cout = 256, NT = Cout / 16;
    int wid  = threadIdx.x >> 5;
    int lane = threadIdx.x & 31;
    int tile = blockIdx.x * (blockDim.x >> 5) + wid;
    if (tile >= (N_NODES / 16) * NT) return;   // wave-uniform (exact divide)
    int nt = tile % NT, mt = tile / NT;
    int half = lane >> 4;
    int gm = mt * 16 + (lane & 15);            // A-row for this lane
    int gn = nt * 16 + (lane & 15);            // B/D-column for this lane
    float inv = inv_cnt[gm];

    v8f acc_l = {}, acc_r = {}, acc_s = {};
#pragma unroll
    for (int k0 = 0; k0 < Cin; k0 += 32) {
        const float* arow = agg + (size_t)gm * Cin + k0;
        const float* xrow = x   + (size_t)gm * Cin + k0;
        v16bf am = load_a_frag(arow, half, inv);    // mean = agg * 1/deg
        v16bf ax = load_a_frag(xrow, half, 1.0f);
        size_t wo = (size_t)gn * Cin + k0 + 16 * half;  // lane K-range, 32B
        v16bf bl = *(const v16bf*)(Wl + wo);
        v16bf br = *(const v16bf*)(Wr + wo);
        v16bf bs = *(const v16bf*)(Ws + wo);
        acc_l = wmma_bf16(am, bl, acc_l);
        acc_r = wmma_bf16(ax, br, acc_r);
        acc_s = wmma_bf16(ax, bs, acc_s);
    }
    float bn = bias[gn];
#pragma unroll
    for (int r = 0; r < 8; ++r) {
        int grow = mt * 16 + r + 8 * half;     // C/D layout: M = r + 8*half
        float v = fmaxf(acc_l[r] + acc_r[r] + bn, 0.0f) + acc_s[r];
        h1[(size_t)grow * Cout + gn] = v;
    }
}

// Layer 2: out = relu(mean2 @ Wl^T + h1 @ Wr^T + b) + h1      (256 -> 256)
__global__ void __launch_bounds__(256)
gemm_layer2(const float* __restrict__ agg, const float* __restrict__ inv_cnt,
            const float* __restrict__ h1, const __bf16* __restrict__ Wl,
            const __bf16* __restrict__ Wr, const float* __restrict__ bias,
            float* __restrict__ out) {
    const int Cin = 256, Cout = 256, NT = Cout / 16;
    int wid  = threadIdx.x >> 5;
    int lane = threadIdx.x & 31;
    int tile = blockIdx.x * (blockDim.x >> 5) + wid;
    if (tile >= (N_NODES / 16) * NT) return;
    int nt = tile % NT, mt = tile / NT;
    int half = lane >> 4;
    int gm = mt * 16 + (lane & 15);
    int gn = nt * 16 + (lane & 15);
    float inv = inv_cnt[gm];

    v8f acc_l = {}, acc_r = {};
#pragma unroll
    for (int k0 = 0; k0 < Cin; k0 += 32) {
        const float* arow = agg + (size_t)gm * Cin + k0;
        const float* hrow = h1  + (size_t)gm * Cin + k0;
        v16bf am = load_a_frag(arow, half, inv);
        v16bf ah = load_a_frag(hrow, half, 1.0f);
        size_t wo = (size_t)gn * Cin + k0 + 16 * half;
        v16bf bl = *(const v16bf*)(Wl + wo);
        v16bf br = *(const v16bf*)(Wr + wo);
        acc_l = wmma_bf16(am, bl, acc_l);
        acc_r = wmma_bf16(ah, br, acc_r);
    }
    float bn = bias[gn];
#pragma unroll
    for (int r = 0; r < 8; ++r) {
        int grow = mt * 16 + r + 8 * half;
        size_t o = (size_t)grow * Cout + gn;
        out[o] = fmaxf(acc_l[r] + acc_r[r] + bn, 0.0f) + h1[o];  // identity skip
    }
}

// ----------------------------- host launcher -------------------------------

extern "C" void kernel_launch(void* const* d_in, const int* in_sizes, int n_in,
                              void* d_out, int out_size, void* d_ws,
                              size_t ws_size, hipStream_t stream) {
    const float*     x   = (const float*)d_in[0];
    const long long* ei  = (const long long*)d_in[1];   // int64 [2, E]
    const float*     Wl1 = (const float*)d_in[2];
    const float*     Wr1 = (const float*)d_in[3];
    const float*     b1  = (const float*)d_in[4];
    const float*     Ws1 = (const float*)d_in[5];
    const float*     Wl2 = (const float*)d_in[6];
    const float*     Wr2 = (const float*)d_in[7];
    const float*     b2  = (const float*)d_in[8];
    float* out = (float*)d_out;

    const int N = N_NODES, E = N_EDGES, C0 = 128, C1 = 256;
    const long long* src = ei;
    const long long* dst = ei + E;

    // Workspace carve-up (256B aligned blocks).
    char*  w   = (char*)d_ws;
    size_t off = 0;
    auto carve = [&](size_t bytes) {
        void* p = w + off;
        off = (off + bytes + 255) & ~(size_t)255;
        return p;
    };
    float*  cnt  = (float*)carve((size_t)N * 4);            // deg -> 1/deg
    float*  agg  = (float*)carve((size_t)N * C1 * 4);       // reused both layers
    float*  h1   = (float*)carve((size_t)N * C1 * 4);
    __bf16* Wl1b = (__bf16*)carve((size_t)C1 * C0 * 2);
    __bf16* Wr1b = (__bf16*)carve((size_t)C1 * C0 * 2);
    __bf16* Ws1b = (__bf16*)carve((size_t)C1 * C0 * 2);
    __bf16* Wl2b = (__bf16*)carve((size_t)C1 * C1 * 2);
    __bf16* Wr2b = (__bf16*)carve((size_t)C1 * C1 * 2);
    (void)ws_size; (void)in_sizes; (void)n_in;

    // Zero accumulators (capturable memset nodes).
    hipMemsetAsync(cnt, 0, (size_t)N * 4, stream);
    hipMemsetAsync(agg, 0, (size_t)N * C1 * 4, stream);

    // Weights -> bf16 once per call.
    int nw1 = C1 * C0, nw2 = C1 * C1;
    cvt_bf16_kernel<<<(nw1 + 255) / 256, 256, 0, stream>>>(Wl1, Wl1b, nw1);
    cvt_bf16_kernel<<<(nw1 + 255) / 256, 256, 0, stream>>>(Wr1, Wr1b, nw1);
    cvt_bf16_kernel<<<(nw1 + 255) / 256, 256, 0, stream>>>(Ws1, Ws1b, nw1);
    cvt_bf16_kernel<<<(nw2 + 255) / 256, 256, 0, stream>>>(Wl2, Wl2b, nw2);
    cvt_bf16_kernel<<<(nw2 + 255) / 256, 256, 0, stream>>>(Wr2, Wr2b, nw2);

    // Degrees and reciprocal (shared by both layers).
    count_kernel<<<(E + 255) / 256, 256, 0, stream>>>(dst, cnt, E);
    inv_kernel<<<(N + 255) / 256, 256, 0, stream>>>(cnt, N);

    // ---- layer 1 ----
    scatter_kernel<128><<<E / 8, 256, 0, stream>>>(x, src, dst, agg, E);
    gemm_layer1<<<(N / 16) * (C1 / 16) / 8, 256, 0, stream>>>(
        agg, cnt, x, Wl1b, Wr1b, Ws1b, b1, h1);

    // ---- layer 2 ----
    hipMemsetAsync(agg, 0, (size_t)N * C1 * 4, stream);
    scatter_kernel<256><<<E / 8, 256, 0, stream>>>(h1, src, dst, agg, E);
    gemm_layer2<<<(N / 16) * (C1 / 16) / 8, 256, 0, stream>>>(
        agg, cnt, h1, Wl2b, Wr2b, b2, out);

    // Tuple output: pass edge_index through after h (raw bytes, D2D copy).
    size_t h_bytes = (size_t)N * C1 * 4;
    size_t e_bytes = (size_t)2 * E * 8;
    if ((size_t)out_size * 4 >= h_bytes + e_bytes) {
        hipMemcpyAsync((char*)d_out + h_bytes, (const void*)ei, e_bytes,
                       hipMemcpyDeviceToDevice, stream);
    }
}
